// Gemma4MonolithicWrapper_62921270886442
// MI455X (gfx1250) — matrix operand order repaired and checked by reading, not compiled, not run on hardware
//
#include <hip/hip_runtime.h>
#include <hip/hip_bf16.h>
#include <math.h>

typedef __attribute__((ext_vector_type(16))) _Float16 v16h;
typedef __attribute__((ext_vector_type(8)))  float    v8f;

constexpr int kH = 2048, kL = 20, kNH = 8, kNKV = 2, kDPL = 256;
constexpr int kDFF = 4096, kV = 16384, kS = 2048, kMAXHD = 512;

static __device__ __forceinline__ v8f wmma16(v16h a, v16h b, v8f c) {
  // D = A(16x32 f16) * B(32x16 f16) + C(16x16 f32)
  return __builtin_amdgcn_wmma_f32_16x16x32_f16(false, a, false, b, (short)0, c,
                                                false, false);
}

// load 8 consecutive f32 (16B-aligned) -> 8 f16 elements of a v16h
static __device__ __forceinline__ void cvt8(const float* __restrict__ p,
                                            v16h& v, const int base) {
  const float4 a = *(const float4*)p;
  const float4 b = *(const float4*)(p + 4);
  v[base + 0] = (_Float16)a.x; v[base + 1] = (_Float16)a.y;
  v[base + 2] = (_Float16)a.z; v[base + 3] = (_Float16)a.w;
  v[base + 4] = (_Float16)b.x; v[base + 5] = (_Float16)b.y;
  v[base + 6] = (_Float16)b.z; v[base + 7] = (_Float16)b.w;
}

// ---------------- elementwise / embedding ----------------

__global__ void k_embed(const int* __restrict__ ids,
                        const float* __restrict__ et, float* __restrict__ h) {
  int j = blockIdx.x * blockDim.x + threadIdx.x;
  if (j < kH) h[j] = et[(size_t)ids[0] * kH + j] * 45.25483399593904f; // sqrt(H)
}

__global__ void k_gelu_mul(const float* __restrict__ a,
                           const float* __restrict__ b, float* __restrict__ o,
                           int n) {
  int i = blockIdx.x * blockDim.x + threadIdx.x;
  if (i >= n) return;
  float x = a[i];
  float t = 0.7978845608028654f * (x + 0.044715f * x * x * x);
  o[i] = 0.5f * x * (1.f + tanhf(t)) * b[i];
}

// rope in-place: rows x hd, pair (d, d+hd/2) handled by one thread
__global__ void k_rope(float* __restrict__ x, const float* __restrict__ cost,
                       const float* __restrict__ sint,
                       const int* __restrict__ posp, int hd) {
  int r = blockIdx.y;
  int d = blockIdx.x * blockDim.x + threadIdx.x;
  int h2 = hd >> 1;
  if (d >= h2) return;
  int pos = posp[0];
  float c = cost[(size_t)pos * hd + d];
  float s = sint[(size_t)pos * hd + d];
  float* xp = x + (size_t)r * hd;
  float x1 = xp[d], x2 = xp[d + h2];
  xp[d]      = x1 * c - x2 * s;
  xp[d + h2] = x2 * c + x1 * s;
}

// ---------------- rmsnorm family ----------------

// y[row] = x[row] * rsqrt(mean(x^2)+eps) * (1+w)   (w==null -> vnorm)
__global__ void k_rms_rows(const float* __restrict__ x,
                           const float* __restrict__ w, float* __restrict__ y,
                           int n) {
  __shared__ float red[256];
  __shared__ float inv;
  int row = blockIdx.x;
  const float* xr = x + (size_t)row * n;
  float* yr = y + (size_t)row * n;
  float acc = 0.f;
  for (int i = threadIdx.x; i < n; i += 256) { float v = xr[i]; acc += v * v; }
  red[threadIdx.x] = acc;
  __syncthreads();
  for (int s = 128; s > 0; s >>= 1) {
    if (threadIdx.x < s) red[threadIdx.x] += red[threadIdx.x + s];
    __syncthreads();
  }
  if (threadIdx.x == 0) inv = rsqrtf(red[0] / (float)n + 1e-6f);
  __syncthreads();
  for (int i = threadIdx.x; i < n; i += 256) {
    float g = w ? (1.f + w[i]) : 1.f;
    yr[i] = xr[i] * inv * g;
  }
}

// hio = (hio + rmsnorm(x,w)) * (*scal or 1)
__global__ void k_add_rms(const float* __restrict__ x,
                          const float* __restrict__ w, float* __restrict__ hio,
                          int n, const float* __restrict__ scal) {
  __shared__ float red[256];
  __shared__ float inv;
  float acc = 0.f;
  for (int i = threadIdx.x; i < n; i += 256) { float v = x[i]; acc += v * v; }
  red[threadIdx.x] = acc;
  __syncthreads();
  for (int s = 128; s > 0; s >>= 1) {
    if (threadIdx.x < s) red[threadIdx.x] += red[threadIdx.x + s];
    __syncthreads();
  }
  if (threadIdx.x == 0) inv = rsqrtf(red[0] / (float)n + 1e-6f);
  __syncthreads();
  float sc = scal ? scal[0] : 1.f;
  for (int i = threadIdx.x; i < n; i += 256)
    hio[i] = (hio[i] + x[i] * inv * (1.f + w[i])) * sc;
}

// combined[l] = (rmsnorm(pl_proj_row, wn) + pl_raw*sqrt(DPL)) / sqrt(2)
__global__ void k_combined(const float* __restrict__ plproj,
                           const float* __restrict__ wn,
                           const float* __restrict__ epl,
                           const int* __restrict__ ids,
                           float* __restrict__ combined) {
  __shared__ float red[256];
  __shared__ float inv;
  int l = blockIdx.x, d = threadIdx.x;
  float v = plproj[l * kDPL + d];
  red[d] = v * v;
  __syncthreads();
  for (int s = 128; s > 0; s >>= 1) {
    if (d < s) red[d] += red[d + s];
    __syncthreads();
  }
  if (d == 0) inv = rsqrtf(red[0] / (float)kDPL + 1e-6f);
  __syncthreads();
  float y = v * inv * (1.f + wn[d]);
  float raw = epl[(size_t)ids[0] * (kL * kDPL) + l * kDPL + d] * 16.f;
  combined[l * kDPL + d] = (y + raw) * 0.7071067811865476f;
}

// ---------------- GEMV: out[n] = scale * sum_k x[k]*W[k*N+n] ----------------
// Each thread produces 4 consecutive outputs: one b128 load per k per thread,
// fully coalesced along n. x staged through LDS. K is a multiple of 256,
// N a multiple of 4.
__global__ void k_gemv(const float* __restrict__ x, const float* __restrict__ W,
                       float* __restrict__ out, int K, int N, float scale) {
  __shared__ float xs[256];
  int n0 = (blockIdx.x * 256 + threadIdx.x) * 4;
  float4 acc = make_float4(0.f, 0.f, 0.f, 0.f);
  for (int k0 = 0; k0 < K; k0 += 256) {
    __syncthreads();
    xs[threadIdx.x] = x[k0 + threadIdx.x];
    __syncthreads();
    if (n0 < N) {
      const float* Wp = W + (size_t)k0 * N + n0;
      if (k0 + 256 < K)
        __builtin_prefetch(W + (size_t)(k0 + 256) * N + n0, 0, 0);
#pragma unroll 8
      for (int kk = 0; kk < 256; kk++) {
        float xv = xs[kk];
        const float4 w = *(const float4*)(Wp + (size_t)kk * N);
        acc.x = fmaf(xv, w.x, acc.x);
        acc.y = fmaf(xv, w.y, acc.y);
        acc.z = fmaf(xv, w.z, acc.z);
        acc.w = fmaf(xv, w.w, acc.w);
      }
    }
  }
  if (n0 < N) {
    acc.x *= scale; acc.y *= scale; acc.z *= scale; acc.w *= scale;
    *(float4*)(out + n0) = acc;
  }
}

// ---------------- attention (WMMA) ----------------
// scores: grid (S/16, NKV), 32 threads. One 16-key tile per wave.
// A rows 0..3 = the 4 query heads of this KV group; rows 4..15 replicate
// (their D rows are never read). B = K-cache tile; the lane whose key == POS
// reads the freshly-computed k row via a single hoisted pointer select.
__global__ void k_attn_scores(const float* __restrict__ q,
                              const float* __restrict__ Kc,
                              const float* __restrict__ kcur,
                              const float* __restrict__ mask,
                              float* __restrict__ scores,
                              const int* __restrict__ posp, int hd,
                              float scale) {
  int kv = blockIdx.y;
  int key0 = blockIdx.x * 16;
  int lane = threadIdx.x;
  int arow = lane & 15;
  int half = lane >> 4;
  int pos = posp[0];
  int s = key0 + (lane & 15);
  const float* Kb = Kc + (size_t)kv * kS * kMAXHD;
  const float* brow = (s == pos) ? (kcur + (size_t)kv * hd)
                                 : (Kb + (size_t)s * kMAXHD);
  const float* aro = q + (size_t)(kv * 4 + (arow & 3)) * hd;
  v8f c = {};
  for (int kb = 0; kb < hd; kb += 32) {
    v16h a, b;
    // A 16x32 f16 layout: lanes 0-15 K={0..7,16..23}, lanes 16-31 K={8..15,24..31}
    cvt8(aro + kb + half * 8, a, 0);
    cvt8(aro + kb + 16 + half * 8, a, 8);
    // B 32x16 f16 layout: lanes 0-15 K={0..7,16..23}, lanes 16-31 K={8..15,24..31}
    cvt8(brow + kb + half * 8, b, 0);
    cvt8(brow + kb + 16 + half * 8, b, 8);
    c = wmma16(a, b, c);
  }
  if (lane < 16) {
    float m = mask[key0 + lane];
#pragma unroll
    for (int r = 0; r < 4; r++)
      scores[(size_t)(kv * 4 + r) * kS + key0 + lane] = c[r] * scale + m;
  }
}

__global__ void k_softmax(float* __restrict__ scores) {
  __shared__ float red[256];
  __shared__ float bc;
  int h = blockIdx.x;
  float* p = scores + (size_t)h * kS;
  float m = -1e30f;
  for (int i = threadIdx.x; i < kS; i += 256) m = fmaxf(m, p[i]);
  red[threadIdx.x] = m;
  __syncthreads();
  for (int s = 128; s > 0; s >>= 1) {
    if (threadIdx.x < s)
      red[threadIdx.x] = fmaxf(red[threadIdx.x], red[threadIdx.x + s]);
    __syncthreads();
  }
  if (threadIdx.x == 0) bc = red[0];
  __syncthreads();
  float M = bc, sum = 0.f;
  for (int i = threadIdx.x; i < kS; i += 256) {
    float e = __expf(p[i] - M);
    p[i] = e;
    sum += e;
  }
  red[threadIdx.x] = sum;
  __syncthreads();
  for (int s = 128; s > 0; s >>= 1) {
    if (threadIdx.x < s) red[threadIdx.x] += red[threadIdx.x + s];
    __syncthreads();
  }
  if (threadIdx.x == 0) bc = 1.f / red[0];
  __syncthreads();
  float r = bc;
  for (int i = threadIdx.x; i < kS; i += 256) p[i] *= r;
}

// ao = probs @ V : grid (hd/16, NKV), 32 threads; 16 output dims per wave.
// V column (fixed d per lane) is strided; the POS-row override is a branchless
// value blend against a loop-invariant vcur[d].
__global__ void k_attn_av(const float* __restrict__ scores,
                          const float* __restrict__ Vc,
                          const float* __restrict__ vcur,
                          float* __restrict__ ao,
                          const int* __restrict__ posp, int hd) {
  int kv = blockIdx.y;
  int d0 = blockIdx.x * 16;
  int lane = threadIdx.x;
  int arow = lane & 15;
  int half = lane >> 4;
  int pos = posp[0];
  int d = d0 + (lane & 15);
  const float* Vb = Vc + (size_t)kv * kS * kMAXHD;
  const float vpos = vcur[(size_t)kv * hd + d];   // loop-invariant
  const float* prow = scores + (size_t)(kv * 4 + (arow & 3)) * kS;
  v8f c = {};
  for (int s0 = 0; s0 < kS; s0 += 32) {
    v16h a, b;
    cvt8(prow + s0 + half * 8, a, 0);
    cvt8(prow + s0 + 16 + half * 8, a, 8);
#pragma unroll
    for (int e = 0; e < 16; e++) {
      int ss = s0 + e + half * 8 + (e >= 8 ? 8 : 0);
      float bv = Vb[(size_t)ss * kMAXHD + d];
      b[e] = (_Float16)((ss == pos) ? vpos : bv);  // v_cndmask, no branch
    }
    c = wmma16(a, b, c);
  }
  if (lane < 16) {
#pragma unroll
    for (int r = 0; r < 4; r++)
      ao[(size_t)(kv * 4 + r) * hd + d0 + lane] = c[r];
  }
}

// ---------------- final argmax with softcap ----------------
__global__ void k_argmax(const float* __restrict__ logits,
                         float* __restrict__ out) {
  __shared__ float vs[256];
  __shared__ int ix[256];
  float best = -1e30f;
  int bi = 0;
  for (int i = threadIdx.x; i < kV; i += 256) {
    float l = tanhf(logits[i] * (1.f / 30.f)) * 30.f;
    if (l > best) { best = l; bi = i; }
  }
  vs[threadIdx.x] = best;
  ix[threadIdx.x] = bi;
  __syncthreads();
  for (int s = 128; s > 0; s >>= 1) {
    if (threadIdx.x < s) {
      float vo = vs[threadIdx.x + s];
      int io = ix[threadIdx.x + s];
      if (vo > vs[threadIdx.x] ||
          (vo == vs[threadIdx.x] && io < ix[threadIdx.x])) {
        vs[threadIdx.x] = vo;
        ix[threadIdx.x] = io;
      }
    }
    __syncthreads();
  }
  if (threadIdx.x == 0) {
    out[0] = (float)ix[0];
    out[1] = vs[0];
  }
}

// ---------------- host orchestration ----------------

extern "C" void kernel_launch(void* const* d_in, const int* in_sizes, int n_in,
                              void* d_out, int out_size, void* d_ws,
                              size_t ws_size, hipStream_t stream) {
  (void)in_sizes; (void)n_in; (void)out_size; (void)ws_size;
  auto F = [&](int i) { return (const float*)d_in[i]; };
  const int* ids = (const int*)d_in[0];
  const int* pos = (const int*)d_in[1];
  const float* mask = F(2);            // (1,1,1,S)
  const float* embed = F(4);
  const float* epl = F(5);
  const float* plw = F(6);
  const float* plnw = F(7);
  const float* cos_s = F(8), *sin_s = F(9), *cos_f = F(10), *sin_f = F(11);
  const float* kvc = F(12);

  struct LP {
    const float *in_ln, *q_w, *q_norm, *o_w, *post_attn_ln, *pre_ff_ln;
    const float *gate_w, *up_w, *down_w, *post_ff_ln;
    const float *pl_gate_w, *pl_proj_w, *post_pl_ln, *layer_scalar;
    const float *k_w, *v_w, *k_norm;
  } lp[kL];
  int t = 13;
  for (int li = 0; li < kL; li++) {
    LP& p = lp[li];
    p.in_ln = F(t++); p.q_w = F(t++); p.q_norm = F(t++); p.o_w = F(t++);
    p.post_attn_ln = F(t++); p.pre_ff_ln = F(t++); p.gate_w = F(t++);
    p.up_w = F(t++); p.down_w = F(t++); p.post_ff_ln = F(t++);
    p.pl_gate_w = F(t++); p.pl_proj_w = F(t++); p.post_pl_ln = F(t++);
    p.layer_scalar = F(t++);
    if (li < kL - 5) { p.k_w = F(t++); p.v_w = F(t++); p.k_norm = F(t++); }
    else { p.k_w = nullptr; p.v_w = nullptr; p.k_norm = nullptr; }
  }
  const float* norm_w = F(t++);
  const float* lmh = F(t++);

  float* ws = (float*)d_ws;
  float* h       = ws;                       // H
  float* hn      = h + kH;                   // H
  float* plproj  = hn + kH;                  // L*DPL
  float* comb    = plproj + kL * kDPL;       // L*DPL
  float* q       = comb + kL * kDPL;         // NH*MAXHD
  float* curk    = q + kNH * kMAXHD;         // NKV*MAXHD
  float* curv    = curk + kNKV * kMAXHD;     // NKV*MAXHD
  float* k13     = curv + kNKV * kMAXHD;     // NKV*256
  float* v13     = k13 + kNKV * 256;         // NKV*256
  float* k14     = v13 + kNKV * 256;         // NKV*512
  float* v14     = k14 + kNKV * kMAXHD;      // NKV*512
  float* scores  = v14 + kNKV * kMAXHD;      // NH*S
  float* ao      = scores + kNH * kS;        // NH*MAXHD
  float* t1      = ao + kNH * kMAXHD;        // V (largest temp: logits)
  float* t2      = t1 + kV;                  // DFF
  float* t3      = t2 + kDFF;                // H

  auto gemv = [&](const float* x, const float* W, float* o, int K, int N,
                  float sc) {
    k_gemv<<<dim3((N + 1023) / 1024), 256, 0, stream>>>(x, W, o, K, N, sc);
  };

  // embedding + per-layer combined
  k_embed<<<kH / 256, 256, 0, stream>>>(ids, embed, h);
  gemv(h, plw, plproj, kH, kL * kDPL, 0.022097086912079608f); // H^-0.5
  k_combined<<<kL, 256, 0, stream>>>(plproj, plnw, epl, ids, comb);

  for (int li = 0; li < kL; li++) {
    const LP& p = lp[li];
    bool full = ((li + 1) % 5) == 0;
    bool shared = li >= kL - 5;
    int hd = full ? 512 : 256;
    const float* ct = full ? cos_f : cos_s;
    const float* st = full ? sin_f : sin_s;
    float scale = full ? 0.04419417382415922f : 0.0625f; // hd^-0.5

    // attention input norm + Q
    k_rms_rows<<<1, 256, 0, stream>>>(h, p.in_ln, hn, kH);
    gemv(hn, p.q_w, q, kH, kNH * hd, 1.f);
    k_rms_rows<<<kNH, 256, 0, stream>>>(q, p.q_norm, q, hd);
    k_rope<<<dim3(1, kNH), 256, 0, stream>>>(q, ct, st, pos, hd);

    const float *kbase, *vbase, *kcur, *vcur;
    if (!shared) {
      gemv(hn, p.k_w, curk, kH, kNKV * hd, 1.f);
      k_rms_rows<<<kNKV, 256, 0, stream>>>(curk, p.k_norm, curk, hd);
      k_rope<<<dim3(1, kNKV), 256, 0, stream>>>(curk, ct, st, pos, hd);
      gemv(hn, p.v_w, curv, kH, kNKV * hd, 1.f);
      k_rms_rows<<<kNKV, 256, 0, stream>>>(curv, nullptr, curv, hd);
      if (li == 13) {
        hipMemcpyAsync(k13, curk, kNKV * 256 * sizeof(float),
                       hipMemcpyDeviceToDevice, stream);
        hipMemcpyAsync(v13, curv, kNKV * 256 * sizeof(float),
                       hipMemcpyDeviceToDevice, stream);
      }
      if (li == 14) {
        hipMemcpyAsync(k14, curk, kNKV * 512 * sizeof(float),
                       hipMemcpyDeviceToDevice, stream);
        hipMemcpyAsync(v14, curv, kNKV * 512 * sizeof(float),
                       hipMemcpyDeviceToDevice, stream);
      }
      kbase = kvc + (size_t)li * kNKV * kS * kMAXHD;
      vbase = kvc + (size_t)(kL + li) * kNKV * kS * kMAXHD;
      kcur = curk; vcur = curv;
    } else {
      int src = full ? 14 : 13;
      kbase = kvc + (size_t)src * kNKV * kS * kMAXHD;
      vbase = kvc + (size_t)(kL + src) * kNKV * kS * kMAXHD;
      kcur = full ? k14 : k13;
      vcur = full ? v14 : v13;
    }

    // attention core (WMMA f16->f32)
    k_attn_scores<<<dim3(kS / 16, kNKV), 32, 0, stream>>>(
        q, kbase, kcur, mask, scores, pos, hd, scale);
    k_softmax<<<kNH, 256, 0, stream>>>(scores);
    k_attn_av<<<dim3(hd / 16, kNKV), 32, 0, stream>>>(scores, vbase, vcur, ao,
                                                      pos, hd);
    gemv(ao, p.o_w, t1, kNH * hd, kH, 1.f);
    k_add_rms<<<1, 256, 0, stream>>>(t1, p.post_attn_ln, h, kH, nullptr);

    // MLP
    k_rms_rows<<<1, 256, 0, stream>>>(h, p.pre_ff_ln, hn, kH);
    gemv(hn, p.gate_w, t1, kH, kDFF, 1.f);
    gemv(hn, p.up_w, t2, kH, kDFF, 1.f);
    k_gelu_mul<<<kDFF / 256, 256, 0, stream>>>(t1, t2, t1, kDFF);
    gemv(t1, p.down_w, t3, kDFF, kH, 1.f);
    k_add_rms<<<1, 256, 0, stream>>>(t3, p.post_ff_ln, h, kH, nullptr);

    // per-layer embedding path
    gemv(h, p.pl_gate_w, t1, kH, kDPL, 1.f);
    k_gelu_mul<<<1, 256, 0, stream>>>(t1, comb + li * kDPL, t1, kDPL);
    gemv(t1, p.pl_proj_w, t3, kDPL, kH, 1.f);
    k_add_rms<<<1, 256, 0, stream>>>(t3, p.post_pl_ln, h, kH, p.layer_scalar);
  }

  // final norm + lm head + softcap argmax
  k_rms_rows<<<1, 256, 0, stream>>>(h, norm_w, hn, kH);
  gemv(hn, lmh, t1, kH, kV, 1.f);
  k_argmax<<<1, 256, 0, stream>>>(t1, (float*)d_out);
}
